// Onet_out_27178553049684
// MI455X (gfx1250) — matrix operand order repaired and checked by reading, not compiled, not run on hardware
//
#include <hip/hip_runtime.h>
#include <hip/hip_bf16.h>
#include <math.h>

typedef __attribute__((ext_vector_type(16))) _Float16 v16h;
typedef __attribute__((ext_vector_type(8)))  float    v8f;

// ---- Kernel-1 LDS layout (bytes). Regions reused across barriered phases.
#define K1_P1   0        // pool1 f16 529*32 = 33856 ; later pool2 (12800), pool3 (2048), flat @4096
#define K1_IN   33856    // input f32 27648 (dead before conv2-out is written over it)
#define K1_C2   33856    // conv2 f16 441*64 = 56448 -> ends 90304 ; later conv3/conv4 weights
#define K1_W2   90304    // conv2 weights f16 288*64 = 36864 -> ends 127168
#define K1_P2   0        // pool2 f16 100*64 = 12800
#define K1_C3   12800    // conv3 f16 64*64 = 8192 -> ends 20992
#define K1_W3   33856    // conv3 weights f16 576*64 = 73728 -> ends 107584
#define K1_P3   0        // pool3 f16 16*64 = 2048
#define K1_W4   33856    // conv4 weights f16 256*128 = 65536 -> ends 99392
#define K1_FL   4096     // flat f16 1152 = 2304 -> ends 6400
#define K1_SMEM 127168

#define K2_SMEM (64*256*2 + 64*16*4)   // dense-out f16 [64][256] + heads f32 [64][16]

#if defined(__HIP_DEVICE_COMPILE__)
// Convert global f32 weights [K][NOUT] -> LDS f16 [co][K] (B-fragment friendly:
// two runs of 8 consecutive K per half-lane -> 8 aligned 32-bit LDS loads per fragment).
__device__ __forceinline__ void stage_weights_f16(_Float16* __restrict__ dst,
                                                  const float* __restrict__ src,
                                                  int K, int NOUT, int tid)
{
  for (int i = tid; i < K * NOUT; i += 256) {   // coalesced global reads
    const int k  = i / NOUT;
    const int co = i - k * NOUT;
    dst[co * K + k] = (_Float16)src[i];
  }
}

// Implicit-GEMM WMMA tile: activations f16 LDS [pos][C], weights f16 LDS [co][KTOT].
// One 16(M=positions) x 16(N=out-channels) f32 accumulator tile.
template<int C, int W, int OUTW, int KH, int KW>
__device__ __forceinline__ v8f wmma_conv_tile(const _Float16* __restrict__ act,
                                              const _Float16* __restrict__ wgt,
                                              int mt, int nt, int lane)
{
  const int lh = lane >> 4;           // lane half selects K offset
  const int ln = lane & 15;           // A: m ; B/D: n
  const int co = nt * 16 + ln;
  const int KTOT = KH * KW * C;
  const int NPOS = OUTW * OUTW;

  int p = mt * 16 + ln;               // clamp invalid M rows (D rows discarded)
  if (p >= NPOS) p = NPOS - 1;
  const int oh = p / OUTW, ow = p - oh * OUTW;
  const _Float16* wrow = wgt + co * KTOT;

  v8f acc = {};
  for (int kb = 0; kb < KTOT; kb += 32) {
    // A: element e -> k = kb + (e/8)*16 + lh*8 + e%8 ; channel pairs -> b32 loads
    union { v16h h; unsigned int u[8]; } A;
#pragma unroll
    for (int j = 0; j < 8; ++j) {
      const int k   = kb + ((j >> 2) << 4) + (lh << 3) + ((j & 3) << 1);
      const int kh  = k / (KW * C);
      const int rem = k - kh * (KW * C);
      const int kw  = rem / C;
      const int ci  = rem - kw * C;   // even
      A.u[j] = *(const unsigned int*)(act + ((oh + kh) * W + (ow + kw)) * C + ci);
    }
    // B: element e -> k = kb + (e/8)*16 + lh*8 + e%8 ; pairs in [co][k] -> b32 loads
    union { v16h h; unsigned int u[8]; } B;
#pragma unroll
    for (int j = 0; j < 8; ++j)
      B.u[j] = *(const unsigned int*)(wrow + kb + ((j >> 2) << 4) + (lh << 3) + ((j & 3) << 1));
    acc = __builtin_amdgcn_wmma_f32_16x16x32_f16(false, A.h, false, B.h,
                                                 (short)0, acc, false, false);
  }
  return acc;
}
#endif

// ==================== Kernel 1: conv stack, one image per block ====================
__global__ __launch_bounds__(256)
void onet_conv_kernel(const float* __restrict__ inputs,
    const float* __restrict__ w1, const float* __restrict__ b1, const float* __restrict__ a1,
    const float* __restrict__ w2, const float* __restrict__ b2, const float* __restrict__ a2,
    const float* __restrict__ w3, const float* __restrict__ b3, const float* __restrict__ a3,
    const float* __restrict__ w4, const float* __restrict__ b4, const float* __restrict__ a4,
    _Float16* __restrict__ flat_out)            // [n_img][1152], permuted (C,W,H)
{
#if defined(__HIP_DEVICE_COMPILE__)
  __shared__ __align__(16) char sm[K1_SMEM];
  const int tid  = threadIdx.x;
  const int img  = blockIdx.x;
  const int lane = tid & 31;
  const int wid  = tid >> 5;
  const int lh   = lane >> 4;
  const int ln   = lane & 15;

  float*    s_in = (float*)(sm + K1_IN);
  _Float16* s_p1 = (_Float16*)(sm + K1_P1);
  _Float16* s_c2 = (_Float16*)(sm + K1_C2);
  _Float16* s_w2 = (_Float16*)(sm + K1_W2);
  _Float16* s_p2 = (_Float16*)(sm + K1_P2);
  _Float16* s_c3 = (_Float16*)(sm + K1_C3);
  _Float16* s_w3 = (_Float16*)(sm + K1_W3);
  _Float16* s_p3 = (_Float16*)(sm + K1_P3);
  _Float16* s_w4 = (_Float16*)(sm + K1_W4);
  _Float16* s_fl = (_Float16*)(sm + K1_FL);

  // ---- Phase 0: stage input image ----
  const float* gin = inputs + (size_t)img * (48 * 48 * 3);
  for (int i = tid; i < 48 * 48 * 3; i += 256) s_in[i] = gin[i];
  __syncthreads();

  // ---- Phase 1: conv1(3x3x3->32)+PReLU+maxpool3x3/s2 (SAME pad 0,1);
  //               also stage conv2 weights into the free top region ----
  {
    const int c = lane;               // lane = channel -> LDS act reads broadcast
    float wr[27];
#pragma unroll
    for (int i = 0; i < 27; ++i) wr[i] = w1[i * 32 + c];
    const float bias = b1[c], al = a1[c];
    for (int pos = wid; pos < 529; pos += 8) {
      const int ph = pos / 23, pw = pos - ph * 23;
      float mx = -3.0e38f;
      for (int dy = 0; dy < 3; ++dy) {
        const int h = ph * 2 + dy; if (h >= 46) break;
        for (int dx = 0; dx < 3; ++dx) {
          const int w = pw * 2 + dx; if (w >= 46) break;
          float s = bias;
#pragma unroll
          for (int kh = 0; kh < 3; ++kh)
#pragma unroll
            for (int kw = 0; kw < 3; ++kw) {
              const float* px = s_in + ((h + kh) * 48 + (w + kw)) * 3;
              const float* wk = wr + (kh * 3 + kw) * 3;
              s = fmaf(px[0], wk[0], s);
              s = fmaf(px[1], wk[1], s);
              s = fmaf(px[2], wk[2], s);
            }
          s = (s >= 0.f) ? s : al * s;
          mx = fmaxf(mx, s);
        }
      }
      s_p1[pos * 32 + c] = (_Float16)mx;
    }
    stage_weights_f16(s_w2, w2, 288, 64, tid);   // region K1_W2 is free
  }
  __syncthreads();

  // ---- Phase 2: conv2 (M=441,N=64,K=288) WMMA + bias + PReLU ----
  for (int t = wid; t < 28 * 4; t += 8) {
    const int mt = t % 28, nt = t / 28;
    v8f acc = wmma_conv_tile<32, 23, 21, 3, 3>(s_p1, s_w2, mt, nt, lane);
    const int co = nt * 16 + ln;
    const float bias = b2[co], al = a2[co];
    if (mt < 27) {                       // uniform full-tile fast path
#pragma unroll
      for (int v = 0; v < 8; ++v) {
        const int p = mt * 16 + v + (lh << 3);
        float x = acc[v] + bias;
        x = (x >= 0.f) ? x : al * x;
        s_c2[p * 64 + co] = (_Float16)x;
      }
    } else {
#pragma unroll
      for (int v = 0; v < 8; ++v) {
        const int p = mt * 16 + v + (lh << 3);
        if (p < 441) {
          float x = acc[v] + bias;
          x = (x >= 0.f) ? x : al * x;
          s_c2[p * 64 + co] = (_Float16)x;
        }
      }
    }
  }
  __syncthreads();

  // ---- Phase 3: maxpool 3x3/s2 VALID: 21x21x64 -> 10x10x64 ----
  for (int o = tid; o < 6400; o += 256) {
    const int c = o & 63, pp = o >> 6;
    const int py = pp / 10, px = pp - py * 10;
    float mx = -3.0e38f;
#pragma unroll
    for (int dy = 0; dy < 3; ++dy)
#pragma unroll
      for (int dx = 0; dx < 3; ++dx)
        mx = fmaxf(mx, (float)s_c2[((py * 2 + dy) * 21 + (px * 2 + dx)) * 64 + c]);
    s_p2[o] = (_Float16)mx;
  }
  __syncthreads();

  // ---- Phase 4a: stage conv3 weights over the dead conv2-out region ----
  stage_weights_f16(s_w3, w3, 576, 64, tid);
  __syncthreads();

  // ---- Phase 4b: conv3 (M=64,N=64,K=576) WMMA + bias + PReLU ----
  for (int t = wid; t < 16; t += 8) {
    const int mt = t & 3, nt = t >> 2;
    v8f acc = wmma_conv_tile<64, 10, 8, 3, 3>(s_p2, s_w3, mt, nt, lane);
    const int co = nt * 16 + ln;
    const float bias = b3[co], al = a3[co];
#pragma unroll
    for (int v = 0; v < 8; ++v) {
      const int p = mt * 16 + v + (lh << 3);   // always < 64
      float x = acc[v] + bias;
      x = (x >= 0.f) ? x : al * x;
      s_c3[p * 64 + co] = (_Float16)x;
    }
  }
  __syncthreads();

  // ---- Phase 5: maxpool 2x2/s2 (8x8x64 -> 4x4x64); stage conv4 weights ----
  for (int o = tid; o < 1024; o += 256) {
    const int c = o & 63, pp = o >> 6;
    const int py = pp >> 2, px = pp & 3;
    float mx = -3.0e38f;
#pragma unroll
    for (int dy = 0; dy < 2; ++dy)
#pragma unroll
      for (int dx = 0; dx < 2; ++dx)
        mx = fmaxf(mx, (float)s_c3[((py * 2 + dy) * 8 + (px * 2 + dx)) * 64 + c]);
    s_p3[o] = (_Float16)mx;
  }
  stage_weights_f16(s_w4, w4, 256, 128, tid);  // over dead conv3-weight region
  __syncthreads();

  // ---- Phase 6: conv4 (M=9->16,N=128,K=256) WMMA; write permuted flat ----
  {
    const int nt = wid;                        // 8 N-tiles, one per wave
    v8f acc = wmma_conv_tile<64, 4, 3, 2, 2>(s_p3, s_w4, 0, nt, lane);
    const int co = nt * 16 + ln;
    const float bias = b4[co], al = a4[co];
#pragma unroll
    for (int v = 0; v < 8; ++v) {
      const int p = v + (lh << 3);
      if (p < 9) {
        const int h = p / 3, w = p - h * 3;
        float x = acc[v] + bias;
        x = (x >= 0.f) ? x : al * x;
        s_fl[co * 9 + w * 3 + h] = (_Float16)x; // (H,W,C) -> (C,W,H) flatten
      }
    }
  }
  __syncthreads();

  // ---- Phase 7: flat -> global scratch (dword-packed, coalesced) ----
  {
    unsigned int* dst = (unsigned int*)(flat_out + (size_t)img * 1152);
    const unsigned int* src = (const unsigned int*)s_fl;
    for (int i = tid; i < 576; i += 256) dst[i] = src[i];
  }
#endif
}

// ========== Kernel 2: batched dense GEMM (M=n_img,K=1152,N=256) + heads ==========
__global__ __launch_bounds__(256)
void dense_heads_kernel(const _Float16* __restrict__ flat,   // [n_img][1152]
    const float* __restrict__ w5, const float* __restrict__ b5, const float* __restrict__ a5,
    const float* __restrict__ wc, const float* __restrict__ bc,
    const float* __restrict__ wb, const float* __restrict__ bb,
    const float* __restrict__ wp, const float* __restrict__ bp,
    float* __restrict__ out, int n_img)
{
#if defined(__HIP_DEVICE_COMPILE__)
  __shared__ __align__(16) char sm[K2_SMEM];
  _Float16* s_d5 = (_Float16*)sm;              // [64][256]
  float*    s_hd = (float*)(sm + 64 * 256 * 2);// [64][16]

  const int tid   = threadIdx.x;
  const int lane  = tid & 31;
  const int wid   = tid >> 5;
  const int lh    = lane >> 4;
  const int ln    = lane & 15;
  const int rbase = blockIdx.x * 64;

  // 64 rows x 256 cols per block: 4 M-tiles x 16 N-tiles = 64 tiles, 8/wave.
  for (int t = wid; t < 64; t += 8) {
    const int mt = t & 3, nt = t >> 2;
    int r = rbase + mt * 16 + ln;
    if (r >= n_img) r = n_img - 1;             // clamp gather; rows unused below
    const _Float16* arow = flat + (size_t)r * 1152;
    const int co = nt * 16 + ln;

    v8f acc = {};
    for (int kb = 0; kb < 1152; kb += 32) {
      union { v16h h; unsigned int u[8]; } A;
#pragma unroll
      for (int j = 0; j < 8; ++j) {
        const int k = kb + ((j >> 2) << 4) + (lh << 3) + ((j & 3) << 1);
        A.u[j] = *(const unsigned int*)(arow + k);
      }
      union { v16h h; _Float16 e[16]; } B;
#pragma unroll
      for (int e = 0; e < 16; ++e)
        B.e[e] = (_Float16)w5[(kb + ((e >> 3) << 4) + (lh << 3) + (e & 7)) * 256 + co];
      acc = __builtin_amdgcn_wmma_f32_16x16x32_f16(false, A.h, false, B.h,
                                                   (short)0, acc, false, false);
    }
    const float bias = b5[co], al = a5[co];
#pragma unroll
    for (int v = 0; v < 8; ++v) {
      const int m = mt * 16 + v + (lh << 3);
      float x = acc[v] + bias;
      x = (x >= 0.f) ? x : al * x;
      s_d5[m * 256 + co] = (_Float16)x;        // PReLU'd dense activations
    }
  }
  __syncthreads();

  // Heads: 64 rows x 16 outputs (2 cls / 4 box / 10 pts), K=256 dots.
  for (int idx = tid; idx < 64 * 16; idx += 256) {
    const int r = idx >> 4, j = idx & 15;
    const float* wm; const float* bm; int cols, jj;
    if (j < 2)      { wm = wc; bm = bc; cols = 2;  jj = j; }
    else if (j < 6) { wm = wb; bm = bb; cols = 4;  jj = j - 2; }
    else            { wm = wp; bm = bp; cols = 10; jj = j - 6; }
    float acc = bm[jj];
    const _Float16* xr = s_d5 + r * 256;
    for (int k = 0; k < 256; ++k)
      acc = fmaf((float)xr[k], wm[k * cols + jj], acc);
    s_hd[idx] = acc;
  }
  __syncthreads();

  // Softmax on cls + write all outputs, one row per thread.
  if (tid < 64) {
    const int img = rbase + tid;
    if (img < n_img) {
      const float* h = s_hd + tid * 16;
      const float m  = fmaxf(h[0], h[1]);
      const float e0 = __expf(h[0] - m), e1 = __expf(h[1] - m);
      const float inv = 1.f / (e0 + e1);
      out[(size_t)img * 2 + 0] = e0 * inv;
      out[(size_t)img * 2 + 1] = e1 * inv;
      float* ob = out + (size_t)n_img * 2 + (size_t)img * 4;
#pragma unroll
      for (int q = 0; q < 4; ++q) ob[q] = h[2 + q];
      float* op = out + (size_t)n_img * 6 + (size_t)img * 10;
#pragma unroll
      for (int q = 0; q < 10; ++q) op[q] = h[6 + q];
    }
  }
#endif
}

extern "C" void kernel_launch(void* const* d_in, const int* in_sizes, int n_in,
                              void* d_out, int out_size, void* d_ws, size_t ws_size,
                              hipStream_t stream) {
  const float* inputs = (const float*)d_in[0];
  const float* w1 = (const float*)d_in[1];
  const float* b1 = (const float*)d_in[2];
  const float* a1 = (const float*)d_in[3];
  const float* w2 = (const float*)d_in[4];
  const float* b2 = (const float*)d_in[5];
  const float* a2 = (const float*)d_in[6];
  const float* w3 = (const float*)d_in[7];
  const float* b3 = (const float*)d_in[8];
  const float* a3 = (const float*)d_in[9];
  const float* w4 = (const float*)d_in[10];
  const float* b4 = (const float*)d_in[11];
  const float* a4 = (const float*)d_in[12];
  const float* w5 = (const float*)d_in[13];
  const float* b5 = (const float*)d_in[14];
  const float* a5 = (const float*)d_in[15];
  const float* wc = (const float*)d_in[16];
  const float* bc = (const float*)d_in[17];
  const float* wb = (const float*)d_in[18];
  const float* bb = (const float*)d_in[19];
  const float* wp = (const float*)d_in[20];
  const float* bp = (const float*)d_in[21];

  const int n_img = in_sizes[0] / (48 * 48 * 3);     // 32*128 = 4096
  _Float16* flat_ws = (_Float16*)d_ws;               // n_img*1152 f16 = 9.4 MB

  onet_conv_kernel<<<n_img, 256, 0, stream>>>(
      inputs, w1, b1, a1, w2, b2, a2, w3, b3, a3, w4, b4, a4, flat_ws);

  dense_heads_kernel<<<(n_img + 63) / 64, 256, 0, stream>>>(
      flat_ws, w5, b5, a5, wc, bc, wb, bb, wp, bp, (float*)d_out, n_img);
}
